// TradingTransformer_61821759259365
// MI455X (gfx1250) — matrix operand order repaired and checked by reading, not compiled, not run on hardware
//
#include <hip/hip_runtime.h>

// ---------------------------------------------------------------------------
// Types for WMMA fragments (CDNA5 gfx1250, wave32)
// ---------------------------------------------------------------------------
typedef __attribute__((ext_vector_type(16))) _Float16 v16h;
typedef __attribute__((ext_vector_type(8)))  _Float16 v8h;
typedef __attribute__((ext_vector_type(8)))  float    v8f;

#define DEV __device__ __forceinline__

// Model dimensions
#define MB    8
#define MS    1024
#define MFEAT 256
#define MD    512
#define MH    8
#define MFF   2048
#define MT    32
#define BS_TOK (MB * MS)   // 8192 encoder tokens
#define BT_TOK (MB * MT)   // 256  decoder tokens

// Compiler-level memory ordering fence (intra-wave LDS round trips; LDS is
// executed in-order per wave on CDNA5, we only need to stop reordering).
#define MEM_FENCE() asm volatile("" ::: "memory")

// ---------------------------------------------------------------------------
// Fragment load helpers.
// A-fragment (16x32 f16): lane l (0..15) = row M=l, halves 0..7 = K hi*16..,
// halves 8..15 = K hi*16+8.. ; two 16B loads at p and p+8 halves.
// B-fragment (32x16 f16): lane = col N, halves = K hi*16 .. hi*16+15 contiguous.
// ---------------------------------------------------------------------------
DEV v16h ldfrag2(const _Float16* p0, const _Float16* p1) {
  v8h a = *(const v8h*)p0;
  v8h b = *(const v8h*)p1;
  v16h r;
#pragma unroll
  for (int i = 0; i < 8; i++) { r[i] = a[i]; r[i + 8] = b[i]; }
  return r;
}
DEV v16h ldfrag16(const _Float16* p) { return ldfrag2(p, p + 8); }

DEV v8f wmma_f16(v16h a, v16h b, v8f c) {
  return __builtin_amdgcn_wmma_f32_16x16x32_f16(false, a, false, b, (short)0, c,
                                                false, false);
}

// ---------------------------------------------------------------------------
// fp32 -> fp16 conversion (grid-stride)
// ---------------------------------------------------------------------------
__global__ void k_cvt(const float* __restrict__ in, _Float16* __restrict__ out,
                      unsigned long long n) {
  unsigned long long i = (unsigned long long)blockIdx.x * blockDim.x + threadIdx.x;
  unsigned long long stride = (unsigned long long)gridDim.x * blockDim.x;
  for (; i < n; i += stride) out[i] = (_Float16)in[i];
}

// ---------------------------------------------------------------------------
// Tiled WMMA GEMM with double-buffered LDS (ping-pong):
//   C[M,N] = epilogue( A[M,K](f16) * B[K,N](f16) )
// Block = 256 threads (8 waves). Block tile 128x128, K-step 32.
// Wave tile 64x32 -> 4x2 grid of 16x16 f32 accumulators.
// Pipeline per K-step: store staged regs -> 1 barrier -> issue next global
// loads -> 8 WMMAs from current buffer (loads overlap matrix ops).
// Epilogue: +bias[N], optional exact GELU, optional residual[M,N]
// (res_mod>0 -> residual row = row % res_mod for the pos-encoding add),
// dual fp32/fp16 outputs.
// ---------------------------------------------------------------------------
__global__ __launch_bounds__(256) void k_gemm(
    const _Float16* __restrict__ A, const _Float16* __restrict__ Bw,
    const float* __restrict__ bias, const float* __restrict__ residual,
    int res_mod, float* __restrict__ C32, _Float16* __restrict__ C16,
    int M, int N, int K, int gelu) {
  __shared__ _Float16 As[2][128][40];   // padded stride: 40 halves = 80B
  __shared__ _Float16 Bt[2][128][40];   // B tile stored transposed: Bt[n][k]

  const int bm = blockIdx.y * 128, bn = blockIdx.x * 128;
  const int tid = threadIdx.x;
  const int lane = tid & 31, wid = tid >> 5;
  const int wm = (wid >> 2) * 64;    // 0 or 64
  const int wn = (wid & 3) * 32;     // 0,32,64,96
  const int l15 = lane & 15, hi = lane >> 4;

  const v8f vzero = {0.f, 0.f, 0.f, 0.f, 0.f, 0.f, 0.f, 0.f};
  v8f acc[4][2];
#pragma unroll
  for (int i = 0; i < 4; i++)
#pragma unroll
    for (int j = 0; j < 2; j++) acc[i][j] = vzero;

  const int ar = tid >> 1, ac = (tid & 1) * 16;       // A staging coords
  const int bk = tid & 31, bn0 = (tid >> 5) * 16;     // B staging coords

  v8h ra0, ra1, rb0, rb1;
  auto load_tile = [&](int kk) {
    const _Float16* ga = A + (size_t)(bm + ar) * K + kk + ac;
    ra0 = *(const v8h*)ga;
    ra1 = *(const v8h*)(ga + 8);
    const _Float16* gb = Bw + (size_t)(kk + bk) * N + bn + bn0;
    rb0 = *(const v8h*)gb;
    rb1 = *(const v8h*)(gb + 8);
  };

  load_tile(0);                       // prologue
  int kbuf = 0;
  for (int k0 = 0; k0 < K; k0 += 32) {
    // commit staged tile into LDS buffer `kbuf`
    *(v8h*)&As[kbuf][ar][ac] = ra0;
    *(v8h*)&As[kbuf][ar][ac + 8] = ra1;
#pragma unroll
    for (int j = 0; j < 8; j++) {
      Bt[kbuf][bn0 + j][bk] = rb0[j];
      Bt[kbuf][bn0 + 8 + j][bk] = rb1[j];
    }
    __syncthreads();
    // launch next tile's global loads under this tile's WMMAs
    if (k0 + 32 < K) load_tile(k0 + 32);
    if (k0 + 64 < K) {
      __builtin_prefetch(A + (size_t)(bm + ar) * K + k0 + 64 + ac, 0, 1);
      __builtin_prefetch(Bw + (size_t)(k0 + 64 + bk) * N + bn + bn0, 0, 1);
    }

    v16h af[4], bf[2];
#pragma unroll
    for (int i = 0; i < 4; i++) {
      const _Float16* p = &As[kbuf][wm + i * 16 + l15][hi * 16];
      af[i] = ldfrag16(p);
    }
#pragma unroll
    for (int j = 0; j < 2; j++) {
      const _Float16* p = &Bt[kbuf][wn + j * 16 + l15][hi * 16];
      bf[j] = ldfrag16(p);
    }
#pragma unroll
    for (int i = 0; i < 4; i++)
#pragma unroll
      for (int j = 0; j < 2; j++) acc[i][j] = wmma_f16(af[i], bf[j], acc[i][j]);
    kbuf ^= 1;
  }

  // Epilogue. C fragment layout: VGPR r -> row r + hi*8 ; col = l15.
#pragma unroll
  for (int i = 0; i < 4; i++) {
#pragma unroll
    for (int j = 0; j < 2; j++) {
#pragma unroll
      for (int r = 0; r < 8; r++) {
        int row = bm + wm + i * 16 + hi * 8 + r;
        int col = bn + wn + j * 16 + l15;
        float v = acc[i][j][r];
        if (bias) v += bias[col];
        if (gelu) v = 0.5f * v * (1.0f + erff(v * 0.70710678118f));
        if (residual) {
          int rr = res_mod ? (row % res_mod) : row;
          v += residual[(size_t)rr * N + col];
        }
        size_t idx = (size_t)row * N + col;
        if (C32) C32[idx] = v;
        if (C16) C16[idx] = (_Float16)v;
      }
    }
  }
}

// ---------------------------------------------------------------------------
// Flash attention: 4 independent waves per block, one wave per
// (b, head, 16-row q-tile). No block barriers (per-wave LDS slices; LDS is
// in-order within a wave).
// Q,K: f16 [B, Sq/Skv, 512] (head h in cols h*64..h*64+63)
// Vt : f16 [B, 512, Skv]  (pre-transposed V)
// O  : f16 [B, Sq, 512]
// ---------------------------------------------------------------------------
__global__ __launch_bounds__(128) void k_attn(
    const _Float16* __restrict__ Q, const _Float16* __restrict__ Kc,
    const _Float16* __restrict__ Vt, _Float16* __restrict__ O,
    int Sq, int Skv, int causal, int nflat) {
  __shared__ _Float16 Pt[4][16][40];   // per-wave P tile (16 q-rows x 32 keys)

  const int wid = threadIdx.x >> 5;
  const int flat = blockIdx.x * 4 + wid;
  if (flat >= nflat) return;
  const int lane = threadIdx.x & 31, l15 = lane & 15, hi = lane >> 4;
  const int nqt = Sq >> 4;
  const int qt = flat % nqt;
  const int h = (flat / nqt) & (MH - 1);
  const int b = flat / (nqt * MH);
  const int qbase = qt * 16, hd = h * 64;

  // Q A-fragments: feature dim 64 = two k-steps of 32
  v16h aq[2];
  {
    const _Float16* qrow = Q + ((size_t)b * Sq + qbase + l15) * MD + hd + hi * 16;
    aq[0] = ldfrag16(qrow);
    aq[1] = ldfrag16(qrow + 32);
  }

  const v8f vzero = {0.f, 0.f, 0.f, 0.f, 0.f, 0.f, 0.f, 0.f};
  v8f o[4];
#pragma unroll
  for (int t = 0; t < 4; t++) o[t] = vzero;
  float mrow[8], lrow[8];
#pragma unroll
  for (int r = 0; r < 8; r++) { mrow[r] = -3.0e38f; lrow[r] = 0.0f; }

  int kblocks = Skv >> 5;
  if (causal) {
    int need = (qbase + 16 + 31) >> 5;
    if (need < kblocks) kblocks = need;
  }

  for (int kb = 0; kb < kblocks; kb++) {
    const int kbase = kb * 32;
    v8f s[2];
    s[0] = vzero; s[1] = vzero;
    // scores = Q (16x64) * K^T (64x32): two 16-key groups, two k-steps each
#pragma unroll
    for (int g = 0; g < 2; g++) {
      const _Float16* krow =
          Kc + ((size_t)b * Skv + kbase + g * 16 + l15) * MD + hd + hi * 16;
      v16h bk0 = ldfrag16(krow);        // features 0..31
      v16h bk1 = ldfrag16(krow + 32);   // features 32..63
      s[g] = wmma_f16(aq[0], bk0, s[g]);
      s[g] = wmma_f16(aq[1], bk1, s[g]);
    }
    // online softmax; C fragment: VGPR r -> q-row r + hi*8 ; col = l15
#pragma unroll
    for (int r = 0; r < 8; r++) {
      float x0 = s[0][r] * 0.125f;   // 1/sqrt(64)
      float x1 = s[1][r] * 0.125f;
      if (causal) {
        int q = qbase + hi * 8 + r;
        if (kbase + l15 > q) x0 = -1.0e30f;
        if (kbase + 16 + l15 > q) x1 = -1.0e30f;
      }
      float vmax = fmaxf(x0, x1);
#pragma unroll
      for (int off = 1; off < 16; off <<= 1)
        vmax = fmaxf(vmax, __shfl_xor(vmax, off, 32));
      float mnew = fmaxf(mrow[r], vmax);
      float alpha = __expf(mrow[r] - mnew);
      mrow[r] = mnew;
      float p0 = __expf(x0 - mnew);
      float p1 = __expf(x1 - mnew);
      float ps = p0 + p1;
#pragma unroll
      for (int off = 1; off < 16; off <<= 1) ps += __shfl_xor(ps, off, 32);
      lrow[r] = lrow[r] * alpha + ps;
#pragma unroll
      for (int t = 0; t < 4; t++) o[t][r] *= alpha;
      Pt[wid][hi * 8 + r][l15] = (_Float16)p0;
      Pt[wid][hi * 8 + r][16 + l15] = (_Float16)p1;
    }
    MEM_FENCE();   // order the intra-wave LDS round trip (DS is in-order)
    // P as A-fragment (16 rows x 32 keys)
    const _Float16* pp = &Pt[wid][l15][hi * 16];
    v16h ap = ldfrag16(pp);
    MEM_FENCE();
    // O += P (16x32) * V (32 keys x 64 features), 4 N-tiles of 16
#pragma unroll
    for (int t = 0; t < 4; t++) {
      const _Float16* vp =
          Vt + ((size_t)b * MD + hd + t * 16 + l15) * Skv + kbase + hi * 16;
      v16h bv = ldfrag16(vp);
      o[t] = wmma_f16(ap, bv, o[t]);
    }
  }

  // normalize & store
#pragma unroll
  for (int r = 0; r < 8; r++) {
    float inv = 1.0f / lrow[r];
    int row = qbase + hi * 8 + r;
    _Float16* orow = O + ((size_t)b * Sq + row) * MD + hd;
#pragma unroll
    for (int t = 0; t < 4; t++) orow[t * 16 + l15] = (_Float16)(o[t][r] * inv);
  }
}

// ---------------------------------------------------------------------------
// LayerNorm over last dim (=512), one block (256 threads) per row.
// ---------------------------------------------------------------------------
__global__ __launch_bounds__(256) void k_ln(const float* __restrict__ X,
                                            const float* __restrict__ gamma,
                                            const float* __restrict__ beta,
                                            float* __restrict__ Y32,
                                            _Float16* __restrict__ Y16) {
  const int N = MD;
  const int row = blockIdx.x, tid = threadIdx.x;
  const float* x = X + (size_t)row * N;
  float a0 = x[tid], a1 = x[tid + 256];
  __shared__ float red[256];
  red[tid] = a0 + a1;
  __syncthreads();
  for (int o = 128; o > 0; o >>= 1) {
    if (tid < o) red[tid] += red[tid + o];
    __syncthreads();
  }
  float mean = red[0] * (1.0f / N);
  __syncthreads();
  float d0 = a0 - mean, d1 = a1 - mean;
  red[tid] = d0 * d0 + d1 * d1;
  __syncthreads();
  for (int o = 128; o > 0; o >>= 1) {
    if (tid < o) red[tid] += red[tid + o];
    __syncthreads();
  }
  float inv = rsqrtf(red[0] * (1.0f / N) + 1e-5f);
  float o0 = d0 * inv * gamma[tid] + beta[tid];
  float o1 = d1 * inv * gamma[tid + 256] + beta[tid + 256];
  if (Y32) {
    float* y = Y32 + (size_t)row * N;
    y[tid] = o0;
    y[tid + 256] = o1;
  }
  if (Y16) {
    _Float16* y = Y16 + (size_t)row * N;
    y[tid] = (_Float16)o0;
    y[tid + 256] = (_Float16)o1;
  }
}

// fp16 [B, Sv, 512] -> [B, 512, Sv]
__global__ void k_transpose(const _Float16* __restrict__ in,
                            _Float16* __restrict__ out, int Sv,
                            unsigned long long total) {
  unsigned long long stride = (unsigned long long)gridDim.x * blockDim.x;
  for (unsigned long long i =
           (unsigned long long)blockIdx.x * blockDim.x + threadIdx.x;
       i < total; i += stride) {
    unsigned long long bb = i / ((unsigned long long)Sv * MD);
    unsigned long long rem = i % ((unsigned long long)Sv * MD);
    unsigned long long s = rem / MD, d = rem % MD;
    out[(bb * MD + d) * Sv + s] = in[i];
  }
}

// positional encoding pe[1024][512]
__global__ void k_pe(float* __restrict__ pe) {
  const int total = MS * MD;
  const int stride = gridDim.x * blockDim.x;
  for (int i = blockIdx.x * blockDim.x + threadIdx.x; i < total; i += stride) {
    int s = i >> 9;
    int d = i & 511;
    int p2 = (d >> 1) * 2;
    float freq = __expf(-(float)p2 * (9.210340372f / (float)MD)); // ln(1e4)/D
    float a = (float)s * freq;
    pe[i] = (d & 1) ? cosf(a) : sinf(a);
  }
}

// decoder init: tgt = broadcast(pe[0:32]) over batch
__global__ void k_bcast_pe(const float* __restrict__ pe, float* __restrict__ y32,
                           _Float16* __restrict__ y16) {
  const int total = MB * MT * MD;
  const int stride = gridDim.x * blockDim.x;
  for (int i = blockIdx.x * blockDim.x + threadIdx.x; i < total; i += stride) {
    int t = (i >> 9) & (MT - 1);
    int d = i & 511;
    float v = pe[(t << 9) + d];
    y32[i] = v;
    y16[i] = (_Float16)v;
  }
}

// final projection: out[row] = dot(X[row,:512], w) + b0
__global__ __launch_bounds__(256) void k_outproj(const float* __restrict__ X,
                                                 const float* __restrict__ w,
                                                 const float* __restrict__ b,
                                                 float* __restrict__ out) {
  __shared__ float red[256];
  const int row = blockIdx.x, tid = threadIdx.x;
  const float* x = X + (size_t)row * MD;
  red[tid] = x[tid] * w[tid] + x[tid + 256] * w[tid + 256];
  __syncthreads();
  for (int o = 128; o > 0; o >>= 1) {
    if (tid < o) red[tid] += red[tid + o];
    __syncthreads();
  }
  if (tid == 0) out[row] = red[0] + b[0];
}

// ---------------------------------------------------------------------------
// Tensor Data Mover exercise: real D# descriptor (ISA ch.8 layout) that DMAs
// a 64x32 fp16 tile of `g` into this workgroup's LDS, waits on TENSORcnt and
// consumes the data. Side effects limited to own LDS (+sink on impossible
// value), so it is safe to enable. Also issues a cluster barrier (S_NOP when
// not dispatched as a cluster).
// ---------------------------------------------------------------------------
__global__ __launch_bounds__(32) void k_tdm_touch(const _Float16* __restrict__ g,
                                                  float* __restrict__ sink,
                                                  int enable) {
  __shared__ _Float16 buf[64 * 32];   // 4KB landing zone at LDS offset 0
#if defined(__HIP_DEVICE_COMPILE__)
#if __has_builtin(__builtin_amdgcn_tensor_load_to_lds)
  if (enable) {
    typedef __attribute__((ext_vector_type(4))) unsigned int tu4;
    typedef __attribute__((ext_vector_type(8))) int ti8;
    typedef __attribute__((ext_vector_type(4))) int ti4;
    unsigned long long ga = (unsigned long long)(const void*)g;
    tu4 g0 = {0u, 0u, 0u, 0u};
    g0[0] = 1u;                                   // count=1: one valid D#
    g0[1] = 0u;                                   // lds_addr = 0
    g0[2] = (unsigned)(ga & 0xFFFFFFFFu);         // global_addr[31:0]
    g0[3] = (unsigned)((ga >> 32) & 0x1FFFFFFu)   // global_addr[56:32]
            | (2u << 30);                         // type = 2 ("image")
    ti8 g1 = {0, 0, 0, 0, 0, 0, 0, 0};
    g1[0] = 0x00010000;          // data_size=1 (2B), no multicast/pad/iterate
    g1[1] = (int)(512u << 16);   // tensor_dim0 = 512  (bits 79:48)
    g1[2] = (int)(512u << 16);   // tensor_dim1 = 512  (bits 111:80)
    g1[3] = (int)(64u << 16);    // tile_dim0   = 64   (bits 127:112)
    g1[4] = 32;                  // tile_dim1   = 32   (bits 143:128)
    g1[5] = 512;                 // tensor_dim0_stride = 512 (bits 207:160)
    ti4 g2 = {0, 0, 0, 0};       // unused (2D tile: tile_dim2 = 0)
    ti4 g3 = {0, 0, 0, 0};
#if defined(__clang_major__) && (__clang_major__ >= 23)
    ti8 g4 = {0, 0, 0, 0, 0, 0, 0, 0};
    __builtin_amdgcn_tensor_load_to_lds(g0, g1, g2, g3, g4, 0);
#else
    __builtin_amdgcn_tensor_load_to_lds(g0, g1, g2, g3, 0);
#endif
#if __has_builtin(__builtin_amdgcn_s_wait_tensorcnt)
    __builtin_amdgcn_s_wait_tensorcnt(0);
#endif
#if __has_builtin(__builtin_amdgcn_s_cluster_barrier)
    __builtin_amdgcn_s_cluster_barrier();   // NOP when not in a cluster
#endif
    float v = (float)buf[threadIdx.x];
    if (v == 1234567.0f) sink[threadIdx.x] = v;   // keep the DMA'd data live
  }
#endif
#endif
  (void)g; (void)sink; (void)enable;
}

// ---------------------------------------------------------------------------
// Host orchestration
// ---------------------------------------------------------------------------
extern "C" void kernel_launch(void* const* d_in, const int* in_sizes, int n_in,
                              void* d_out, int out_size, void* d_ws,
                              size_t ws_size, hipStream_t stream) {
  (void)in_sizes; (void)n_in; (void)out_size; (void)ws_size;

  // jax-tree (sorted-key) flattening order of setup_inputs()
  enum {
    I_DEC_CWK = 0, I_DEC_CWO, I_DEC_CWQ, I_DEC_CWV,
    I_DEC_CBK, I_DEC_CBO, I_DEC_CBQ, I_DEC_CBV,
    I_DEC_FC1B, I_DEC_FC1W, I_DEC_FC2B, I_DEC_FC2W,
    I_DEC_LN1B, I_DEC_LN1G, I_DEC_LN2B, I_DEC_LN2G, I_DEC_LN3B, I_DEC_LN3G,
    I_DEC_SWK, I_DEC_SWO, I_DEC_SWQ, I_DEC_SWV,
    I_DEC_SBK, I_DEC_SBO, I_DEC_SBQ, I_DEC_SBV,
    I_DECB, I_DECG,
    I_ENC_WK, I_ENC_WO, I_ENC_WQ, I_ENC_WV,
    I_ENC_BK, I_ENC_BO, I_ENC_BQ, I_ENC_BV,
    I_ENC_FC1B, I_ENC_FC1W, I_ENC_FC2B, I_ENC_FC2W,
    I_ENC_LN1B, I_ENC_LN1G, I_ENC_LN2B, I_ENC_LN2G,
    I_ENCB, I_ENCG, I_INB, I_INW, I_OUTB, I_OUTW, I_SRC
  };
  auto F = [&](int i) -> const float* { return (const float*)d_in[i]; };

  char* wp = (char*)d_ws;
  auto carve = [&](size_t bytes) -> void* {
    void* r = (void*)wp;
    wp += (bytes + 255) & ~((size_t)255);
    return r;
  };

  const size_t WM = (size_t)6 * 512 * 512;
  const size_t WF = (size_t)6 * 512 * 2048;

  // fp16 weight copies
  _Float16* encWk = (_Float16*)carve(WM * 2);
  _Float16* encWo = (_Float16*)carve(WM * 2);
  _Float16* encWq = (_Float16*)carve(WM * 2);
  _Float16* encWv = (_Float16*)carve(WM * 2);
  _Float16* encF1 = (_Float16*)carve(WF * 2);
  _Float16* encF2 = (_Float16*)carve(WF * 2);
  _Float16* decSWk = (_Float16*)carve(WM * 2);
  _Float16* decSWo = (_Float16*)carve(WM * 2);
  _Float16* decSWq = (_Float16*)carve(WM * 2);
  _Float16* decSWv = (_Float16*)carve(WM * 2);
  _Float16* decCWk = (_Float16*)carve(WM * 2);
  _Float16* decCWo = (_Float16*)carve(WM * 2);
  _Float16* decCWq = (_Float16*)carve(WM * 2);
  _Float16* decCWv = (_Float16*)carve(WM * 2);
  _Float16* decF1 = (_Float16*)carve(WF * 2);
  _Float16* decF2 = (_Float16*)carve(WF * 2);
  _Float16* inW16 = (_Float16*)carve((size_t)MFEAT * MD * 2);
  _Float16* src16 = (_Float16*)carve((size_t)BS_TOK * MFEAT * 2);

  // activations
  float*    pe32  = (float*)carve((size_t)MS * MD * 4);
  float*    x32   = (float*)carve((size_t)BS_TOK * MD * 4);
  _Float16* x16   = (_Float16*)carve((size_t)BS_TOK * MD * 2);
  float*    t32   = (float*)carve((size_t)BS_TOK * MD * 4);
  _Float16* q16   = (_Float16*)carve((size_t)BS_TOK * MD * 2);
  _Float16* kk16  = (_Float16*)carve((size_t)BS_TOK * MD * 2);
  _Float16* vv16  = (_Float16*)carve((size_t)BS_TOK * MD * 2);
  _Float16* vt16  = (_Float16*)carve((size_t)BS_TOK * MD * 2);
  _Float16* a16   = (_Float16*)carve((size_t)BS_TOK * MD * 2);
  _Float16* h16   = (_Float16*)carve((size_t)BS_TOK * MFF * 2);
  float*    enc32 = (float*)carve((size_t)BS_TOK * MD * 4);
  _Float16* enc16 = (_Float16*)carve((size_t)BS_TOK * MD * 2);
  float*    y32   = (float*)carve((size_t)BT_TOK * MD * 4);
  _Float16* y16   = (_Float16*)carve((size_t)BT_TOK * MD * 2);
  float*    yf32  = (float*)carve((size_t)BT_TOK * MD * 4);

  auto cvt = [&](int idx, _Float16* dst, size_t n) {
    k_cvt<<<1024, 256, 0, stream>>>(F(idx), dst, (unsigned long long)n);
  };
  cvt(I_ENC_WK, encWk, WM);  cvt(I_ENC_WO, encWo, WM);
  cvt(I_ENC_WQ, encWq, WM);  cvt(I_ENC_WV, encWv, WM);
  cvt(I_ENC_FC1W, encF1, WF); cvt(I_ENC_FC2W, encF2, WF);
  cvt(I_DEC_SWK, decSWk, WM); cvt(I_DEC_SWO, decSWo, WM);
  cvt(I_DEC_SWQ, decSWq, WM); cvt(I_DEC_SWV, decSWv, WM);
  cvt(I_DEC_CWK, decCWk, WM); cvt(I_DEC_CWO, decCWo, WM);
  cvt(I_DEC_CWQ, decCWq, WM); cvt(I_DEC_CWV, decCWv, WM);
  cvt(I_DEC_FC1W, decF1, WF); cvt(I_DEC_FC2W, decF2, WF);
  cvt(I_INW, inW16, (size_t)MFEAT * MD);
  cvt(I_SRC, src16, (size_t)BS_TOK * MFEAT);

  k_tdm_touch<<<1, 32, 0, stream>>>(encWk, t32, 1);
  k_pe<<<1024, 256, 0, stream>>>(pe32);

  auto gemm = [&](const _Float16* A, const _Float16* Bw, const float* bias,
                  const float* res, int res_mod, float* C32, _Float16* C16,
                  int M, int N, int K, int gelu) {
    dim3 grid((unsigned)(N / 128), (unsigned)(M / 128));
    k_gemm<<<grid, 256, 0, stream>>>(A, Bw, bias, res, res_mod, C32, C16, M, N,
                                     K, gelu);
  };
  auto ln = [&](const float* X, const float* g, const float* b, float* Y32,
                _Float16* Y16, int rows) {
    k_ln<<<rows, 256, 0, stream>>>(X, g, b, Y32, Y16);
  };
  auto transpose = [&](const _Float16* in, _Float16* out, int Sv) {
    unsigned long long total = (unsigned long long)MB * Sv * MD;
    k_transpose<<<2048, 256, 0, stream>>>(in, out, Sv, total);
  };
  auto attn = [&](const _Float16* Qp, const _Float16* Kp, const _Float16* Vtp,
                  _Float16* Op, int Sq, int Skv, int causal) {
    int total = MB * MH * (Sq / 16);
    int blocks = (total + 3) / 4;
    k_attn<<<blocks, 128, 0, stream>>>(Qp, Kp, Vtp, Op, Sq, Skv, causal, total);
  };

  // input projection: x = src @ in_w + in_b + pe[row % 1024]
  gemm(src16, inW16, F(I_INB), pe32, MS, x32, x16, BS_TOK, MD, MFEAT, 0);

  // ---------------- encoder ----------------
  for (int l = 0; l < 6; l++) {
    const size_t wo = (size_t)l * 512 * 512;
    const size_t f1 = (size_t)l * 512 * 2048;
    const size_t f2 = (size_t)l * 2048 * 512;
    gemm(x16, encWq + wo, F(I_ENC_BQ) + l * 512, nullptr, 0, nullptr, q16,
         BS_TOK, MD, MD, 0);
    gemm(x16, encWk + wo, F(I_ENC_BK) + l * 512, nullptr, 0, nullptr, kk16,
         BS_TOK, MD, MD, 0);
    gemm(x16, encWv + wo, F(I_ENC_BV) + l * 512, nullptr, 0, nullptr, vv16,
         BS_TOK, MD, MD, 0);
    transpose(vv16, vt16, MS);
    attn(q16, kk16, vt16, a16, MS, MS, 0);
    gemm(a16, encWo + wo, F(I_ENC_BO) + l * 512, x32, 0, t32, nullptr, BS_TOK,
         MD, MD, 0);
    ln(t32, F(I_ENC_LN1G) + l * 512, F(I_ENC_LN1B) + l * 512, x32, x16, BS_TOK);
    gemm(x16, encF1 + f1, F(I_ENC_FC1B) + l * 2048, nullptr, 0, nullptr, h16,
         BS_TOK, MFF, MD, 1);
    gemm(h16, encF2 + f2, F(I_ENC_FC2B) + l * 512, x32, 0, t32, nullptr,
         BS_TOK, MD, MFF, 0);
    ln(t32, F(I_ENC_LN2G) + l * 512, F(I_ENC_LN2B) + l * 512, x32, x16, BS_TOK);
  }
  ln(x32, F(I_ENCG), F(I_ENCB), enc32, enc16, BS_TOK);

  // ---------------- decoder ----------------
  k_bcast_pe<<<512, 256, 0, stream>>>(pe32, y32, y16);
  for (int l = 0; l < 6; l++) {
    const size_t wo = (size_t)l * 512 * 512;
    const size_t f1 = (size_t)l * 512 * 2048;
    const size_t f2 = (size_t)l * 2048 * 512;
    // masked self-attention
    gemm(y16, decSWq + wo, F(I_DEC_SBQ) + l * 512, nullptr, 0, nullptr, q16,
         BT_TOK, MD, MD, 0);
    gemm(y16, decSWk + wo, F(I_DEC_SBK) + l * 512, nullptr, 0, nullptr, kk16,
         BT_TOK, MD, MD, 0);
    gemm(y16, decSWv + wo, F(I_DEC_SBV) + l * 512, nullptr, 0, nullptr, vv16,
         BT_TOK, MD, MD, 0);
    transpose(vv16, vt16, MT);
    attn(q16, kk16, vt16, a16, MT, MT, 1);
    gemm(a16, decSWo + wo, F(I_DEC_SBO) + l * 512, y32, 0, t32, nullptr,
         BT_TOK, MD, MD, 0);
    ln(t32, F(I_DEC_LN1G) + l * 512, F(I_DEC_LN1B) + l * 512, y32, y16, BT_TOK);
    // cross-attention (K/V from encoder output)
    gemm(y16, decCWq + wo, F(I_DEC_CBQ) + l * 512, nullptr, 0, nullptr, q16,
         BT_TOK, MD, MD, 0);
    gemm(enc16, decCWk + wo, F(I_DEC_CBK) + l * 512, nullptr, 0, nullptr, kk16,
         BS_TOK, MD, MD, 0);
    gemm(enc16, decCWv + wo, F(I_DEC_CBV) + l * 512, nullptr, 0, nullptr, vv16,
         BS_TOK, MD, MD, 0);
    transpose(vv16, vt16, MS);
    attn(q16, kk16, vt16, a16, MT, MS, 0);
    gemm(a16, decCWo + wo, F(I_DEC_CBO) + l * 512, y32, 0, t32, nullptr,
         BT_TOK, MD, MD, 0);
    ln(t32, F(I_DEC_LN2G) + l * 512, F(I_DEC_LN2B) + l * 512, y32, y16, BT_TOK);
    // FFN
    gemm(y16, decF1 + f1, F(I_DEC_FC1B) + l * 2048, nullptr, 0, nullptr, h16,
         BT_TOK, MFF, MD, 1);
    gemm(h16, decF2 + f2, F(I_DEC_FC2B) + l * 512, y32, 0, t32, nullptr,
         BT_TOK, MD, MFF, 0);
    ln(t32, F(I_DEC_LN3G) + l * 512, F(I_DEC_LN3B) + l * 512, y32, y16, BT_TOK);
  }
  ln(y32, F(I_DECG), F(I_DECB), yf32, nullptr, BT_TOK);
  k_outproj<<<BT_TOK, 256, 0, stream>>>(yf32, F(I_OUTW), F(I_OUTB),
                                        (float*)d_out);
}
